// ActionPreprocess_62423054680471
// MI455X (gfx1250) — matrix operand order repaired and checked by reading, not compiled, not run on hardware
//
#include <hip/hip_runtime.h>
#include <hip/hip_bf16.h>

typedef __attribute__((ext_vector_type(16))) _Float16 v16h;
typedef __attribute__((ext_vector_type(8)))  float    v8f;

#define EMB   64
#define NQ    20
#define HWX   (48 * 48)          // 2304
#define CELLS (16 * HWX)         // 36864
#define KPAD  32                 // 19 live K rows padded to 32

// ---------------------------------------------------------------------------
// Kernel 1: build the f16 B operand (Table, w2 pre-folded) directly in the
// WMMA B-matrix lane layout, one copy per 16-wide N tile:
//   bpack[tile][lane][elem] , lane 0..31, elem 0..15 (halves)
//   column  N = tile*16 + (lane & 15)
//   K index   = (elem < 8) ? 8*(lane>>4) + elem : 16 + 8*(lane>>4) + (elem - 8)
// K row meaning: 0-5 type, 6-10 direction, 11-15 resource, 16-17 repeat,
//                18 = all-ones (carries the per-cell scalar), 19-31 = 0.
// ---------------------------------------------------------------------------
__global__ void build_btable(const float* __restrict__ type_t,
                             const float* __restrict__ dir_t,
                             const float* __restrict__ res_t,
                             const float* __restrict__ rep_t,
                             const float* __restrict__ w2,
                             _Float16* __restrict__ bpack) {
    int idx = blockIdx.x * blockDim.x + threadIdx.x;
    if (idx >= 4 * 32 * 16) return;
    int tile = idx >> 9;
    int lane = (idx >> 4) & 31;
    int elem = idx & 15;
    int n = tile * 16 + (lane & 15);
    int k = (elem < 8) ? ((lane >> 4) * 8 + elem) : (16 + (lane >> 4) * 8 + (elem - 8));
    float v = 0.0f;
    if      (k < 6)   v = w2[0] * type_t[k * EMB + n];
    else if (k < 11)  v = w2[1] * dir_t[(k - 6) * EMB + n];
    else if (k < 16)  v = w2[2] * res_t[(k - 11) * EMB + n];
    else if (k < 18)  v = w2[4] * rep_t[(k - 16) * EMB + n];
    else if (k == 18) v = 1.0f;
    bpack[idx] = (_Float16)v;
}

// ---------------------------------------------------------------------------
// Per-cell weighted-histogram accumulator (all static field access -> regs)
// ---------------------------------------------------------------------------
struct Acc {
    float ct0, ct1, ct2, ct3, ct4, ct5;     // type (6)
    float cd0, cd1, cd2, cd3, cd4;          // direction (5)
    float cr0, cr1, cr2, cr3, cr4;          // resource (5)
    float cp0, cp1;                         // repeat!=0 (2)
    float amtS, nS;                         // scalar contractions
};

__device__ __forceinline__ void accum(Acc& s, int a, int d, int r, int p,
                                      float am, float nv, float wq) {
    s.ct0 += (a == 0) ? wq : 0.0f;
    s.ct1 += (a == 1) ? wq : 0.0f;
    s.ct2 += (a == 2) ? wq : 0.0f;
    s.ct3 += (a == 3) ? wq : 0.0f;
    s.ct4 += (a == 4) ? wq : 0.0f;
    s.ct5 += (a == 5) ? wq : 0.0f;
    s.cd0 += (d == 0) ? wq : 0.0f;
    s.cd1 += (d == 1) ? wq : 0.0f;
    s.cd2 += (d == 2) ? wq : 0.0f;
    s.cd3 += (d == 3) ? wq : 0.0f;
    s.cd4 += (d == 4) ? wq : 0.0f;
    s.cr0 += (r == 0) ? wq : 0.0f;
    s.cr1 += (r == 1) ? wq : 0.0f;
    s.cr2 += (r == 2) ? wq : 0.0f;
    s.cr3 += (r == 3) ? wq : 0.0f;
    s.cr4 += (r == 4) ? wq : 0.0f;
    s.cp0 += (p == 0) ? wq : 0.0f;
    s.cp1 += (p != 0) ? wq : 0.0f;
    s.amtS += wq * am;
    s.nS   += wq * nv;
}

// ---------------------------------------------------------------------------
// Kernel 2: main fused kernel. One wave32 handles 16 consecutive cells:
//   build A (16 cells x K=32, f16, ISA 16-bit A layout), then 4 WMMAs
//   against the 4 packed N tiles; LeakyReLU + unit-mask the f32 accumulator
//   and store transposed [B,E,H,W] with two contiguous float4 per lane.
// ---------------------------------------------------------------------------
__global__ __launch_bounds__(256)
void action_preprocess(const unsigned char* __restrict__ unit,
                       const int*   __restrict__ act_type,
                       const int*   __restrict__ direction,
                       const int*   __restrict__ resource,
                       const int*   __restrict__ repeat_,
                       const float* __restrict__ amount,
                       const float* __restrict__ nbuf,
                       const float* __restrict__ w1,
                       const float* __restrict__ b1,
                       const float* __restrict__ w2,
                       const float* __restrict__ b2,
                       const _Float16* __restrict__ bpack,
                       float* __restrict__ out) {
    const int lane = threadIdx.x & 31;
    const int wave = threadIdx.x >> 5;
    const int base = (blockIdx.x * 8 + wave) * 16;   // first cell of this wave
    if (base >= CELLS) return;                       // wave-uniform; EXEC stays full
    const bool hi = lane >= 16;
    const int  m    = lane & 15;
    const int  cell = base + m;                      // lane pair (L, L+16) share a cell
    const int  qb   = cell * NQ;                     // 80-byte (16B aligned) stride

    // ---- build the 19 coefficients for this cell (branchless, chunked B128) ----
    Acc s = {};
    const int4*   ap  = (const int4*)(act_type  + qb);
    const int4*   dp  = (const int4*)(direction + qb);
    const int4*   rp  = (const int4*)(resource  + qb);
    const int4*   pp  = (const int4*)(repeat_   + qb);
    const float4* amp = (const float4*)(amount + qb);
    const float4* np  = (const float4*)(nbuf   + qb);
    const float4* w1p = (const float4*)w1;
#pragma unroll
    for (int c = 0; c < NQ / 4; ++c) {
        int4   a4 = ap[c], d4 = dp[c], r4 = rp[c], p4 = pp[c];
        float4 m4 = amp[c], n4 = np[c], w4 = w1p[c];
        accum(s, a4.x, d4.x, r4.x, p4.x, m4.x, n4.x, w4.x);
        accum(s, a4.y, d4.y, r4.y, p4.y, m4.y, n4.y, w4.y);
        accum(s, a4.z, d4.z, r4.z, p4.z, m4.z, n4.z, w4.z);
        accum(s, a4.w, d4.w, r4.w, p4.w, m4.w, n4.w, w4.w);
    }
    const float sw2 = w2[0] + w2[1] + w2[2] + w2[3] + w2[4] + w2[5];
    const float scalar = w2[3] * s.amtS + w2[5] * s.nS + b1[0] * sw2 + b2[0];

    // ---- pack A per the ISA 16-bit A layout ----
    // lanes 0-15 : halves 0..7 = K0..7,  halves 8..15 = K16..23
    // lanes 16-31: halves 0..7 = K8..15, halves 8..15 = K24..31
    v16h A;
    A[0]  = (_Float16)(hi ? s.cd2 : s.ct0);
    A[1]  = (_Float16)(hi ? s.cd3 : s.ct1);
    A[2]  = (_Float16)(hi ? s.cd4 : s.ct2);
    A[3]  = (_Float16)(hi ? s.cr0 : s.ct3);
    A[4]  = (_Float16)(hi ? s.cr1 : s.ct4);
    A[5]  = (_Float16)(hi ? s.cr2 : s.ct5);
    A[6]  = (_Float16)(hi ? s.cr3 : s.cd0);
    A[7]  = (_Float16)(hi ? s.cr4 : s.cd1);
    A[8]  = (_Float16)(hi ? 0.0f  : s.cp0);
    A[9]  = (_Float16)(hi ? 0.0f  : s.cp1);
    A[10] = (_Float16)(hi ? 0.0f  : scalar);
    A[11] = (_Float16)0.0f;
    A[12] = (_Float16)0.0f;
    A[13] = (_Float16)0.0f;
    A[14] = (_Float16)0.0f;
    A[15] = (_Float16)0.0f;

    // ---- per-lane unit mask for this lane's 8 M rows (C/D layout: M = r + hi*8) ----
    const unsigned char* up = unit + base + (hi ? 8 : 0);
    float msk[8];
#pragma unroll
    for (int i = 0; i < 8; ++i) msk[i] = up[i] ? 1.0f : 0.0f;

    // ---- output addressing: group never crosses a batch boundary (2304 % 16 == 0) ----
    const int bb  = base / HWX;
    const int hwb = base % HWX;
    float* obase = out + (size_t)bb * (EMB * HWX) + hwb + (hi ? 8 : 0);

#pragma unroll
    for (int t = 0; t < 4; ++t) {
        v16h B = *(const v16h*)(bpack + ((t * 32 + lane) << 4));
        v8f  C = {};
        C = __builtin_amdgcn_wmma_f32_16x16x32_f16(
                /*neg_a=*/false, A, /*neg_b=*/false, B,
                /*c_mod=*/(short)0, C, /*reuse_a=*/false, /*reuse_b=*/false);
        const int e = t * 16 + m;                     // embedding dim for this lane
        float* op = obase + (size_t)e * HWX;
        float4 s0, s1;
#pragma unroll
        for (int r = 0; r < 8; ++r) {
            float v = C[r];
            v = (v >= 0.0f) ? v : 0.01f * v;          // LeakyReLU
            v *= msk[r];                              // unit_indicator mask
            ((float*)&s0)[r < 4 ? r : 0] = v;         // for r >= 4 clobbers s0[0], which is never restored
            if (r < 4) ((float*)&s0)[r] = v; else ((float*)&s1)[r - 4] = v;
        }
        *(float4*)op       = s0;                      // 8 consecutive hw addresses
        *(float4*)(op + 4) = s1;
    }
}

// ---------------------------------------------------------------------------
extern "C" void kernel_launch(void* const* d_in, const int* in_sizes, int n_in,
                              void* d_out, int out_size, void* d_ws, size_t ws_size,
                              hipStream_t stream) {
    const unsigned char* unit   = (const unsigned char*)d_in[0];  // bool -> 1 byte
    const int*   act_type  = (const int*)d_in[1];
    const int*   direction = (const int*)d_in[2];
    const int*   resource  = (const int*)d_in[3];
    const int*   repeat_   = (const int*)d_in[4];
    const float* amount    = (const float*)d_in[5];
    const float* nbuf      = (const float*)d_in[6];
    const float* type_t    = (const float*)d_in[7];
    const float* dir_t     = (const float*)d_in[8];
    const float* res_t     = (const float*)d_in[9];
    const float* rep_t     = (const float*)d_in[10];
    const float* w1        = (const float*)d_in[11];
    const float* b1        = (const float*)d_in[12];
    const float* w2        = (const float*)d_in[13];
    const float* b2        = (const float*)d_in[14];
    float* out = (float*)d_out;

    _Float16* bpack = (_Float16*)d_ws;                 // 4 tiles * 32 lanes * 16 halves = 4 KB

    build_btable<<<8, 256, 0, stream>>>(type_t, dir_t, res_t, rep_t, w2, bpack);

    const int blocks = CELLS / 128;                    // 8 waves * 16 cells per block = 288
    action_preprocess<<<blocks, 256, 0, stream>>>(
        unit, act_type, direction, resource, repeat_, amount, nbuf,
        w1, b1, w2, b2, bpack, out);
}
